// MotifsBackend_29334626632039
// MI455X (gfx1250) — compile-verified, matrix operand order checked by reading, not run on hardware
//
#include <hip/hip_runtime.h>

// ---------------------------------------------------------------------------
// Problem dims (fixed by the reference)
// ---------------------------------------------------------------------------
#define L_MOTIF 12
#define B_   8
#define S_   4096
#define C_   64
#define M_   8
#define H2_  512      // hidden width after GEMM1 (= M_*C_ too)
#define H_   256      // output width
#define ROWS (B_ * S_)          // 32768
#define TROWS 16                // row tile per block

typedef __attribute__((ext_vector_type(16))) __bf16       v16bf;
typedef __attribute__((ext_vector_type(8)))  float        v8f;
typedef __attribute__((ext_vector_type(4)))  unsigned int v4u;

union Frag16 { v16bf v; v4u q[2]; };

__device__ __forceinline__ unsigned short f2bf(float f) {
  unsigned int u = __builtin_bit_cast(unsigned int, f);
  u = (u + 0x7FFFu + ((u >> 16) & 1u)) >> 16;   // round-to-nearest-even
  return (unsigned short)u;
}

__device__ __forceinline__ float gelu_exact(float x) {
  return 0.5f * x * (1.0f + erff(x * 0.70710678118654752f));
}

// ---------------------------------------------------------------------------
// Kernel 1: z-normalize motifs (m*c = 512 rows of length 12)
// ---------------------------------------------------------------------------
__global__ void motifz_kernel(const float* __restrict__ motifs,
                              float* __restrict__ mz) {
  int t = blockIdx.x * blockDim.x + threadIdx.x;
  if (t >= M_ * C_) return;
  const float* src = motifs + t * L_MOTIF;
  float v[L_MOTIF], s = 0.f, sq = 0.f;
#pragma unroll
  for (int j = 0; j < L_MOTIF; ++j) { v[j] = src[j]; s += v[j]; sq += v[j] * v[j]; }
  float mean = s * (1.0f / L_MOTIF);
  float var  = sq * (1.0f / L_MOTIF) - mean * mean;
  float inv  = 1.0f / (sqrtf(fmaxf(var, 0.f)) + 1e-8f);
  float* dst = mz + t * L_MOTIF;
#pragma unroll
  for (int j = 0; j < L_MOTIF; ++j) dst[j] = (v[j] - mean) * inv;
}

// ---------------------------------------------------------------------------
// Kernel 2: pack a K x N f32 weight into bf16 WMMA-B fragments.
// Fragment (kt,nt) = 32x16 tile, stored lane-major: 32 lanes x 16 bf16.
// Lane = half*16 + n_local (half = K half of the tile); positions 0..7 hold
// k = half*8 + p, positions 8..15 hold k = 16 + half*8 + (p-8).
// ---------------------------------------------------------------------------
__global__ void pack_b_kernel(const float* __restrict__ W,
                              unsigned short* __restrict__ dst,
                              int K, int N) {
  int e = blockIdx.x * blockDim.x + threadIdx.x;
  int total = K * N;
  if (e >= total) return;
  int frag   = e >> 9;         // 512 elements per fragment
  int within = e & 511;
  int lane = within >> 4;
  int p    = within & 15;
  int ntiles = N >> 4;
  int kt = frag / ntiles;
  int nt = frag - kt * ntiles;
  int half = lane >> 4, nl = lane & 15;
  int kl = (p < 8) ? (half * 8 + p) : (16 + half * 8 + (p - 8));
  int k = kt * 32 + kl;
  int n = nt * 16 + nl;
  dst[e] = f2bf(W[k * N + n]);
}

// ---------------------------------------------------------------------------
// Kernel 3: fused  windows_z -> sim -> relu^4 -> GEMM1(bf16 WMMA) -> gelu ->
//           LayerNorm -> GEMM2(bf16 WMMA) -> out   (one block per 16 rows)
//
// Dynamic LDS layout (bytes):
//   [0      , 24576)  motifs_z  (6144 f32)        | aliased by hn tile (ph4)
//   [24576  , 73728)  windows_z (16*64*12 f32)    | aliased by LN red (ph3)
//   [73728  , 90112)  flat tile (16*512 bf16)
// ---------------------------------------------------------------------------
__global__ __launch_bounds__(256)
void fused_kernel(const float* __restrict__ x,
                  const float* __restrict__ mz,
                  const unsigned short* __restrict__ W1p,
                  const float* __restrict__ b1,
                  const float* __restrict__ gamma,
                  const float* __restrict__ beta,
                  const unsigned short* __restrict__ W2p,
                  const float* __restrict__ b2,
                  float* __restrict__ out) {
  extern __shared__ char smem[];
  float*          mzS   = (float*)(smem);
  float*          wzS   = (float*)(smem + 24576);
  unsigned short* flatS = (unsigned short*)(smem + 73728);
  unsigned short* hnS   = (unsigned short*)(smem);          // aliases mzS
  float*          redS  = (float*)(smem + 24576);           // aliases wzS
  float*          redQ  = redS + 16;

  const int tid  = threadIdx.x;
  const int lane = tid & 31;
  const int wave = tid >> 5;
  const int halfL = lane >> 4;
  const int nl    = lane & 15;
  const int Mrow  = lane & 15;   // A-fragment row for this lane

  const int gr0  = blockIdx.x * TROWS;      // first global row
  const int bIdx = gr0 >> 12;               // 4096 rows per batch
  const int s0   = gr0 & (S_ - 1);

  // -------- phase 0: stage motifs_z ----------------------------------------
  for (int i = tid; i < M_ * C_ * L_MOTIF; i += 256) mzS[i] = mz[i];

  // -------- phase 1: per (row, channel) window z-score ---------------------
  const float* xb = x + (size_t)bIdx * S_ * C_;
  for (int task = tid; task < TROWS * C_; task += 256) {
    int row = task >> 6;
    int c   = task & 63;
    int s   = s0 + row;
    float v[L_MOTIF], sum = 0.f, sq = 0.f;
#pragma unroll
    for (int j = 0; j < L_MOTIF; ++j) {
      int si = s + j - (L_MOTIF - 1);
      si = si < 0 ? 0 : si;                 // left pad = replicate first step
      float val = xb[si * C_ + c];
      v[j] = val; sum += val; sq += val * val;
    }
    float mean = sum * (1.0f / L_MOTIF);
    float var  = sq * (1.0f / L_MOTIF) - mean * mean;
    float inv  = 1.0f / (sqrtf(fmaxf(var, 0.f)) + 1e-8f);
    float* dst = wzS + task * L_MOTIF;
#pragma unroll
    for (int j = 0; j < L_MOTIF; ++j) dst[j] = (v[j] - mean) * inv;
  }
  __syncthreads();

  // -------- phase 2: sim -> relu^4 -> bf16 flat tile -----------------------
  for (int o = tid; o < TROWS * H2_; o += 256) {
    int row = o >> 9;
    int mc  = o & 511;                      // m*64 + c (matches W1 row order)
    const float* wp = wzS + (row * C_ + (mc & 63)) * L_MOTIF;
    const float* mp = mzS + mc * L_MOTIF;
    float d = 0.f;
#pragma unroll
    for (int j = 0; j < L_MOTIF; ++j) d += wp[j] * mp[j];
    float sim = d * (1.0f / L_MOTIF);
    float r = fmaxf(sim, 0.f);
    float g = r * r; g = g * g;             // relu(sim)^4
    flatS[o] = f2bf(g);
  }
  __syncthreads();                          // flat ready; wz/mz now dead

  if (tid < 32) redS[tid] = 0.f;            // zero LN reduction slots
  __syncthreads();

  // -------- phase 3: GEMM1  h1 = flat @ W1 + b1  (bf16 WMMA, f32 acc) ------
  v8f acc1[4];
#pragma unroll
  for (int j = 0; j < 4; ++j) {
    int col = (wave * 4 + j) * 16 + nl;
    float bv = b1[col];
#pragma unroll
    for (int r = 0; r < 8; ++r) acc1[j][r] = bv;
  }
  for (int kt = 0; kt < H2_ / 32; ++kt) {
    Frag16 a;
    const unsigned short* arow = flatS + Mrow * H2_ + kt * 32 + halfL * 8;
    a.q[0] = *(const v4u*)(arow);
    a.q[1] = *(const v4u*)(arow + 16);
    if (kt + 1 < H2_ / 32)
      __builtin_prefetch(W1p + ((size_t)((kt + 1) * 32 + wave * 4) << 9) + lane * 16, 0, 0);
#pragma unroll
    for (int j = 0; j < 4; ++j) {
      int nt = wave * 4 + j;
      const unsigned short* bp = W1p + ((size_t)(kt * 32 + nt) << 9) + lane * 16;
      Frag16 b;
      b.q[0] = *(const v4u*)(bp);
      b.q[1] = *(const v4u*)(bp + 8);
      acc1[j] = __builtin_amdgcn_wmma_f32_16x16x32_bf16(
          false, a.v, false, b.v, (short)0, acc1[j], false, false);
    }
  }

  // gelu + LN partial sums (each lane holds 4 cols x 8 rows)
  float g1[4][8];
  float rsum[8], rq[8];
#pragma unroll
  for (int r = 0; r < 8; ++r) { rsum[r] = 0.f; rq[r] = 0.f; }
#pragma unroll
  for (int j = 0; j < 4; ++j)
#pragma unroll
    for (int r = 0; r < 8; ++r) {
      float h = gelu_exact(acc1[j][r]);
      g1[j][r] = h;
      rsum[r] += h; rq[r] += h * h;
    }
#pragma unroll
  for (int r = 0; r < 8; ++r) {
    int row = r + 8 * halfL;                // C/D layout: lanes 16-31 -> M+8
    atomicAdd(&redS[row], rsum[r]);
    atomicAdd(&redQ[row], rq[r]);
  }
  __syncthreads();

  float muA[8], rsA[8];
#pragma unroll
  for (int r = 0; r < 8; ++r) {
    int row = r + 8 * halfL;
    float mu  = redS[row] * (1.0f / H2_);
    float var = redQ[row] * (1.0f / H2_) - mu * mu;
    muA[r] = mu;
    rsA[r] = rsqrtf(var + 1e-5f);
  }
#pragma unroll
  for (int j = 0; j < 4; ++j) {
    int col = (wave * 4 + j) * 16 + nl;
    float gm = gamma[col], bt = beta[col];
#pragma unroll
    for (int r = 0; r < 8; ++r) {
      int row = r + 8 * halfL;
      float val = (g1[j][r] - muA[r]) * rsA[r] * gm + bt;
      hnS[row * H2_ + col] = f2bf(val);
    }
  }
  __syncthreads();                          // hn tile ready (aliases mz region)

  // -------- phase 4: GEMM2  out = hn @ W2 + b2 -----------------------------
  v8f acc2[2];
#pragma unroll
  for (int j = 0; j < 2; ++j) {
    int col = (wave * 2 + j) * 16 + nl;
    float bv = b2[col];
#pragma unroll
    for (int r = 0; r < 8; ++r) acc2[j][r] = bv;
  }
  for (int kt = 0; kt < H2_ / 32; ++kt) {
    Frag16 a;
    const unsigned short* arow = hnS + Mrow * H2_ + kt * 32 + halfL * 8;
    a.q[0] = *(const v4u*)(arow);
    a.q[1] = *(const v4u*)(arow + 16);
#pragma unroll
    for (int j = 0; j < 2; ++j) {
      int nt = wave * 2 + j;
      const unsigned short* bp = W2p + ((size_t)(kt * (H_ / 16) + nt) << 9) + lane * 16;
      Frag16 b;
      b.q[0] = *(const v4u*)(bp);
      b.q[1] = *(const v4u*)(bp + 8);
      acc2[j] = __builtin_amdgcn_wmma_f32_16x16x32_bf16(
          false, a.v, false, b.v, (short)0, acc2[j], false, false);
    }
  }
#pragma unroll
  for (int j = 0; j < 2; ++j) {
    int col = (wave * 2 + j) * 16 + nl;
#pragma unroll
    for (int r = 0; r < 8; ++r) {
      int row = r + 8 * halfL;
      out[(size_t)(gr0 + row) * H_ + col] = acc2[j][r];
    }
  }
}

// ---------------------------------------------------------------------------
// Launcher.  Workspace needs: 24 KB motifs_z + 512 KB W1 pack + 256 KB W2 pack
// (< 1 MB total).
// ---------------------------------------------------------------------------
extern "C" void kernel_launch(void* const* d_in, const int* in_sizes, int n_in,
                              void* d_out, int out_size, void* d_ws, size_t ws_size,
                              hipStream_t stream) {
  const float* x      = (const float*)d_in[0];
  const float* motifs = (const float*)d_in[1];
  const float* W1     = (const float*)d_in[2];
  const float* b1     = (const float*)d_in[3];
  const float* gamma  = (const float*)d_in[4];
  const float* beta   = (const float*)d_in[5];
  const float* W2     = (const float*)d_in[6];
  const float* b2     = (const float*)d_in[7];
  float* out = (float*)d_out;

  char* ws = (char*)d_ws;
  float*          mz  = (float*)ws;                                   // 24576 B
  unsigned short* W1p = (unsigned short*)(ws + 32768);                // 524288 B
  unsigned short* W2p = (unsigned short*)(ws + 32768 + 524288);       // 262144 B

  motifz_kernel<<<2, 256, 0, stream>>>(motifs, mz);
  pack_b_kernel<<<(H2_ * H2_ + 255) / 256, 256, 0, stream>>>(W1, W1p, H2_, H2_);
  pack_b_kernel<<<(H2_ * H_  + 255) / 256, 256, 0, stream>>>(W2, W2p, H2_, H_);

  const size_t smem = 90112;   // 24K mz + 48K wz + 16K flat (hn/red alias)
  (void)hipFuncSetAttribute((const void*)fused_kernel,
                            hipFuncAttributeMaxDynamicSharedMemorySize,
                            (int)smem);
  fused_kernel<<<ROWS / TROWS, 256, smem, stream>>>(
      x, mz, W1p, b1, gamma, beta, W2p, b2, out);
}
